// DAGLayer_46694884442362
// MI455X (gfx1250) — matrix operand order repaired and checked by reading, not compiled, not run on hardware
//
#include <hip/hip_runtime.h>
#include <hip/hip_bf16.h>

typedef __bf16 v16bf __attribute__((ext_vector_type(16)));
typedef __bf16 v4bf  __attribute__((ext_vector_type(4)));
typedef float  v8f   __attribute__((ext_vector_type(8)));

#define DIM     128
#define LN_EPSF 1e-5f
#define STR     132      // padded LDS row stride (elements)
#define NWAVE   4        // waves per block in main kernel

__device__ __forceinline__ v8f wmma_bf16(v16bf a, v16bf b, v8f c) {
  // D = A(16x32 bf16) x B(32x16 bf16) + C(16x16 f32)
  return __builtin_amdgcn_wmma_f32_16x16x32_bf16(false, a, false, b, (short)0, c, false, false);
}

// Build a 16-element bf16 A-fragment from two runs of 8 consecutive f32s.
__device__ __forceinline__ v16bf cvt16(const float* p0, const float* p1) {
  const float4 x0 = *(const float4*)(p0);
  const float4 x1 = *(const float4*)(p0 + 4);
  const float4 y0 = *(const float4*)(p1);
  const float4 y1 = *(const float4*)(p1 + 4);
  v16bf r;
  r[0]=(__bf16)x0.x;  r[1]=(__bf16)x0.y;  r[2]=(__bf16)x0.z;  r[3]=(__bf16)x0.w;
  r[4]=(__bf16)x1.x;  r[5]=(__bf16)x1.y;  r[6]=(__bf16)x1.z;  r[7]=(__bf16)x1.w;
  r[8]=(__bf16)y0.x;  r[9]=(__bf16)y0.y;  r[10]=(__bf16)y0.z; r[11]=(__bf16)y0.w;
  r[12]=(__bf16)y1.x; r[13]=(__bf16)y1.y; r[14]=(__bf16)y1.z; r[15]=(__bf16)y1.w;
  return r;
}

__global__ void k_zero(float* p, long n) {
  long i = (long)blockIdx.x * blockDim.x + threadIdx.x;
  long stride = (long)gridDim.x * blockDim.x;
  for (; i < n; i += stride) p[i] = 0.f;
}

__global__ void k_cvt_w(const float* __restrict__ Ws, const float* __restrict__ Wn,
                        const float* __restrict__ Wg,
                        __bf16* wsb, __bf16* wnb, __bf16* wg1, __bf16* wg2) {
  int t = blockIdx.x * blockDim.x + threadIdx.x;       // 65536 threads
  if (t < 16384) {
    wsb[t] = (__bf16)Ws[t];
  } else if (t < 32768) {
    int i = t - 16384;
    wnb[i] = (__bf16)Wn[i];
  } else {
    int t2 = t - 32768;                                // W_gate [128][256] row-major
    int j = t2 >> 8, k = t2 & 255;
    __bf16 v = (__bf16)Wg[t2];
    if (k < 128) wg1[j * 128 + k] = v;
    else         wg2[j * 128 + (k - 128)] = v;
  }
}

__global__ void k_scatter(const float* __restrict__ h, const int* __restrict__ src,
                          const int* __restrict__ dst, float* nsum, float* cnt, int nE) {
  long t = (long)blockIdx.x * blockDim.x + threadIdx.x;
  int e = (int)(t >> 5);
  int lane = (int)(t & 31);
  if (e >= nE) return;
  int s = src[e], d = dst[e];
  float4 v = *(const float4*)(h + (size_t)s * DIM + lane * 4);
  float* o = nsum + (size_t)d * DIM + lane * 4;
  atomicAdd(o + 0, v.x); atomicAdd(o + 1, v.y);
  atomicAdd(o + 2, v.z); atomicAdd(o + 3, v.w);
  if (lane == 0) atomicAdd(cnt + d, 1.0f);
}

__global__ void k_finalize(float* nsum, const float* __restrict__ cnt,
                           float* hp, int nNodes) {
  long i = (long)blockIdx.x * blockDim.x + threadIdx.x;
  long n = (long)nNodes * DIM;
  if (i >= n) return;
  int node = (int)(i >> 7);
  float c = cnt[node];
  nsum[i] = nsum[i] / fmaxf(c, 1.f);
  if ((i & (DIM - 1)) == 0) hp[node] = (c > 0.f) ? 1.f : 0.f;
}

__global__ void __launch_bounds__(NWAVE * 32)
k_main(const float* __restrict__ h, const float* __restrict__ nmean,
       const float* __restrict__ hp,
       const __bf16* __restrict__ wsb, const __bf16* __restrict__ wnb,
       const __bf16* __restrict__ wg1, const __bf16* __restrict__ wg2,
       const float* __restrict__ b_self, const float* __restrict__ b_neigh,
       const float* __restrict__ b_gate, const float* __restrict__ ln_g,
       const float* __restrict__ ln_b, float* __restrict__ out,
       int nNodes, int nTiles) {
  __shared__ float  lds[NWAVE * 16 * STR];   // m / v stage (f32)
  __shared__ __bf16 gls[NWAVE * 16 * STR];   // gate stage (bf16)
  const int lane = threadIdx.x & 31;
  const int wid  = threadIdx.x >> 5;
  const int tile = blockIdx.x * NWAVE + wid;
  if (tile >= nTiles) return;                 // wave-uniform exit, no barriers used
  float*  mst = &lds[wid * 16 * STR];
  __bf16* gst = &gls[wid * 16 * STR];
  const int base = tile * 16;
  const int arow = lane & 15;
  const int khA  = (lane < 16) ? 0 : 16;      // A-frag K base (contiguous K, matches B)
  const int khB  = (lane < 16) ? 0 : 16;      // B-frag K base (contiguous K, matches A)
  const int r0   = (lane < 16) ? 0 : 8;       // C/D row group

  int nodeA = base + arow; if (nodeA >= nNodes) nodeA = nNodes - 1;
  const float* hrow = h + (size_t)nodeA * DIM;
  const float* nrow = nmean + (size_t)nodeA * DIM;

  v16bf a_h[4], a_n[4];
#pragma unroll
  for (int kt = 0; kt < 4; ++kt) {
    int k0 = kt * 32 + khA;
    a_h[kt] = cvt16(hrow + k0, hrow + k0 + 8);
    a_n[kt] = cvt16(nrow + k0, nrow + k0 + 8);
  }

  float p8[8];
#pragma unroll
  for (int v = 0; v < 8; ++v) {
    int nr = base + r0 + v; if (nr >= nNodes) nr = nNodes - 1;
    p8[v] = hp[nr];
  }

  // hs / hn GEMMs -> m staged in LDS
  for (int nt = 0; nt < 8; ++nt) {
    const int col = nt * 16 + arow;
    v8f accS = {0.f,0.f,0.f,0.f,0.f,0.f,0.f,0.f};
    v8f accN = accS;
#pragma unroll
    for (int kt = 0; kt < 4; ++kt) {
      v16bf bS = *(const v16bf*)(wsb + col * DIM + kt * 32 + khB);
      accS = wmma_bf16(a_h[kt], bS, accS);
      v16bf bN = *(const v16bf*)(wnb + col * DIM + kt * 32 + khB);
      accN = wmma_bf16(a_n[kt], bN, accN);
    }
    const float bs = b_self[col], bn = b_neigh[col];
#pragma unroll
    for (int v = 0; v < 8; ++v)
      mst[(r0 + v) * STR + col] = (accS[v] + bs) + p8[v] * (accN[v] + bn);
  }

  // reload m as A-fragments (f32 LDS -> bf16)
#pragma unroll
  for (int kt = 0; kt < 4; ++kt) {
    int k0 = kt * 32 + khA;
    a_n[kt] = cvt16(&mst[arow * STR + k0], &mst[arow * STR + k0 + 8]);
  }

  // gate GEMM: logits = h@Wg1^T + m@Wg2^T ; sigmoid -> gate stage (bf16)
  for (int nt = 0; nt < 8; ++nt) {
    const int col = nt * 16 + arow;
    v8f accG = {0.f,0.f,0.f,0.f,0.f,0.f,0.f,0.f};
#pragma unroll
    for (int kt = 0; kt < 4; ++kt) {
      v16bf b1 = *(const v16bf*)(wg1 + col * DIM + kt * 32 + khB);
      accG = wmma_bf16(a_h[kt], b1, accG);
      v16bf b2 = *(const v16bf*)(wg2 + col * DIM + kt * 32 + khB);
      accG = wmma_bf16(a_n[kt], b2, accG);
    }
    const float bg = b_gate[col];
#pragma unroll
    for (int v = 0; v < 8; ++v) {
      float g = 1.f / (1.f + __expf(-(accG[v] + bg)));
      gst[(r0 + v) * STR + col] = (__bf16)g;
    }
  }

  // v = g*m + (1-g)*h, computed row-major with coalesced h loads
  for (int r = 0; r < 16; ++r) {
    int nr = base + r; int nrc = (nr >= nNodes) ? (nNodes - 1) : nr;
    float4 m4 = *(const float4*)&mst[r * STR + lane * 4];
    v4bf  g4b = *(const v4bf*)&gst[r * STR + lane * 4];
    float4 h4 = *(const float4*)(h + (size_t)nrc * DIM + lane * 4);
    float g0 = (float)g4b[0], g1 = (float)g4b[1], g2 = (float)g4b[2], g3 = (float)g4b[3];
    float4 vv;
    vv.x = g0 * m4.x + (1.f - g0) * h4.x;
    vv.y = g1 * m4.y + (1.f - g1) * h4.y;
    vv.z = g2 * m4.z + (1.f - g2) * h4.z;
    vv.w = g3 * m4.w + (1.f - g3) * h4.w;
    *(float4*)&mst[r * STR + lane * 4] = vv;
  }

  // LayerNorm + ReLU: lane pair (r, r+16) reduces the two halves of row r
  float s = 0.f, s2 = 0.f;
  const float* rowp = &mst[arow * STR + ((lane < 16) ? 0 : 64)];
#pragma unroll
  for (int i = 0; i < 16; ++i) {
    float4 x = *(const float4*)(rowp + i * 4);
    s  += x.x + x.y + x.z + x.w;
    s2 += x.x * x.x + x.y * x.y + x.z * x.z + x.w * x.w;
  }
  s  += __shfl_xor(s, 16, 32);
  s2 += __shfl_xor(s2, 16, 32);
  const float mu = s * (1.f / DIM);
  const float rs = rsqrtf(s2 * (1.f / DIM) - mu * mu + LN_EPSF);

  const float4 g4 = *(const float4*)(ln_g + lane * 4);
  const float4 bt = *(const float4*)(ln_b + lane * 4);
  for (int r = 0; r < 16; ++r) {
    float mur = __shfl(mu, r, 32);
    float rsr = __shfl(rs, r, 32);
    int nr = base + r;
    if (nr < nNodes) {
      float4 x = *(const float4*)&mst[r * STR + lane * 4];
      float4 o;
      o.x = fmaxf((x.x - mur) * rsr * g4.x + bt.x, 0.f);
      o.y = fmaxf((x.y - mur) * rsr * g4.y + bt.y, 0.f);
      o.z = fmaxf((x.z - mur) * rsr * g4.z + bt.z, 0.f);
      o.w = fmaxf((x.w - mur) * rsr * g4.w + bt.w, 0.f);
      *(float4*)(out + (size_t)nr * DIM + lane * 4) = o;
    }
  }
}

extern "C" void kernel_launch(void* const* d_in, const int* in_sizes, int n_in,
                              void* d_out, int out_size, void* d_ws, size_t ws_size,
                              hipStream_t stream) {
  const float* h    = (const float*)d_in[0];
  const int*   esrc = (const int*)d_in[1];
  const int*   edst = (const int*)d_in[2];
  const float* Ws   = (const float*)d_in[3];
  const float* bs   = (const float*)d_in[4];
  const float* Wn   = (const float*)d_in[5];
  const float* bn   = (const float*)d_in[6];
  const float* Wg   = (const float*)d_in[7];
  const float* bg   = (const float*)d_in[8];
  const float* lg   = (const float*)d_in[9];
  const float* lb   = (const float*)d_in[10];
  float* out = (float*)d_out;

  const int nNodes = in_sizes[0] / DIM;
  const int nEdges = in_sizes[1];

  // workspace layout
  float* nsum = (float*)d_ws;                       // [nNodes*128]
  float* cnt  = nsum + (size_t)nNodes * DIM;        // [nNodes]
  float* hp   = cnt + nNodes;                       // [nNodes]
  uintptr_t wb = ((uintptr_t)(hp + nNodes) + 31) & ~(uintptr_t)31;
  __bf16* wsb = (__bf16*)wb;                        // 4 x [128*128] bf16
  __bf16* wnb = wsb + DIM * DIM;
  __bf16* wg1 = wnb + DIM * DIM;
  __bf16* wg2 = wg1 + DIM * DIM;

  long zn = (long)nNodes * DIM + nNodes;            // nsum + cnt contiguous
  k_zero<<<2048, 256, 0, stream>>>(nsum, zn);
  k_cvt_w<<<65536 / 256, 256, 0, stream>>>(Ws, Wn, Wg, wsb, wnb, wg1, wg2);
  long st = (long)nEdges * 32;
  k_scatter<<<(int)((st + 255) / 256), 256, 0, stream>>>(h, esrc, edst, nsum, cnt, nEdges);
  long fn = (long)nNodes * DIM;
  k_finalize<<<(int)((fn + 255) / 256), 256, 0, stream>>>(nsum, cnt, hp, nNodes);
  int nTiles = (nNodes + 15) / 16;
  k_main<<<(nTiles + NWAVE - 1) / NWAVE, NWAVE * 32, 0, stream>>>(
      h, nsum, hp, wsb, wnb, wg1, wg2, bs, bn, bg, lg, lb, out, nNodes, nTiles);
}
